// MultiheadAttention_27771258536670
// MI455X (gfx1250) — matrix operand order repaired and checked by reading, not compiled, not run on hardware
//
#include <hip/hip_runtime.h>
#include <hip/hip_bf16.h>

typedef __attribute__((ext_vector_type(16))) __bf16 v16bf;
typedef __attribute__((ext_vector_type(8)))  __bf16 v8bf;
typedef __attribute__((ext_vector_type(8)))  float  v8f;
typedef __attribute__((ext_vector_type(4)))  int    v4i;

#define S_LEN  2048
#define IN_DIM 512
#define OUT_DIM 512
#define NH 16
#define NKV 4
#define HD 64
#define QDIM (NH * HD)    // 1024
#define KVDIM (NKV * HD)  // 256

#define AS1 __attribute__((address_space(1)))
#define AS3 __attribute__((address_space(3)))

typedef AS1 v4i* gbl_v4i_ptr;   // pointer to <4 x i32> in global address space
typedef AS3 v4i* lds_v4i_ptr;   // pointer to <4 x i32> in LDS address space

#if __has_builtin(__builtin_amdgcn_global_load_async_to_lds_b128)
#define HAVE_ASYNC_LDS 1
#else
#define HAVE_ASYNC_LDS 0
#endif

// ---------- helpers ----------

__device__ __forceinline__ __bf16 f2bf(float f) {
  unsigned u = __builtin_bit_cast(unsigned, f);
  unsigned r = u + 0x7FFFu + ((u >> 16) & 1u); // round-to-nearest-even
  unsigned short h = (unsigned short)(r >> 16);
  return __builtin_bit_cast(__bf16, h);
}

// A-matrix 16x32 bf16 fragment (contiguous k order, same as B loader): lane holds row (lane&15);
// lanes 0-15: K = k0..k0+15; lanes 16-31: K = k0+16..k0+31.
__device__ __forceinline__ v16bf load_a16x32(const __bf16* A, int lda, int m0, int k0) {
  int lane = threadIdx.x & 31;
  const __bf16* p = A + (size_t)(m0 + (lane & 15)) * lda + k0 + ((lane >> 4) << 4);
  v8bf lo = *(const v8bf*)p;
  v8bf hi = *(const v8bf*)(p + 8);
  return __builtin_shufflevector(lo, hi, 0,1,2,3,4,5,6,7,8,9,10,11,12,13,14,15);
}

// B-matrix 32x16 bf16 fragment; tile element (k,n) = Bt[n][k] (Bt row-major [N][K]).
// lane holds col (lane&15); lanes 0-15: K=k0..k0+15; lanes 16-31: K=k0+16..k0+31.
__device__ __forceinline__ v16bf load_b32x16(const __bf16* Bt, int ldb, int n0, int k0) {
  int lane = threadIdx.x & 31;
  const __bf16* p = Bt + (size_t)(n0 + (lane & 15)) * ldb + k0 + ((lane >> 4) << 4);
  v8bf lo = *(const v8bf*)p;
  v8bf hi = *(const v8bf*)(p + 8);
  return __builtin_shufflevector(lo, hi, 0,1,2,3,4,5,6,7,8,9,10,11,12,13,14,15);
}

__device__ __forceinline__ v8f wmma_bf16(v16bf a, v16bf b, v8f c) {
  return __builtin_amdgcn_wmma_f32_16x16x32_bf16(false, a, false, b, (short)0, c, false, false);
}

// xor-shuffle within 16-lane groups via ds_swizzle (group-of-32 mode, and=0x1f, or=0)
template<int M>
__device__ __forceinline__ float swz_xor(float v) {
  int i = __builtin_bit_cast(int, v);
  i = __builtin_amdgcn_ds_swizzle(i, (M << 10) | 0x1f);
  return __builtin_bit_cast(float, i);
}
__device__ __forceinline__ float rowmax16(float v) {
  v = fmaxf(v, swz_xor<8>(v)); v = fmaxf(v, swz_xor<4>(v));
  v = fmaxf(v, swz_xor<2>(v)); v = fmaxf(v, swz_xor<1>(v));
  return v;
}
__device__ __forceinline__ float rowsum16(float v) {
  v += swz_xor<8>(v); v += swz_xor<4>(v); v += swz_xor<2>(v); v += swz_xor<1>(v);
  return v;
}

__device__ __forceinline__ void wait_async_lds() {
#if HAVE_ASYNC_LDS
#if __has_builtin(__builtin_amdgcn_s_wait_asynccnt)
  __builtin_amdgcn_s_wait_asynccnt(0);
#else
  asm volatile("s_wait_asynccnt 0x0" ::: "memory");
#endif
#endif
}

// Stage a 64x64 bf16 K-tile (rows t0..t0+63, cols kvoff..kvoff+63 of kb [S, KVDIM])
// into LDS dst (row-major 64x64). 16 issues of 32 lanes x 16B.
__device__ __forceinline__ void stage_k_tile(const __bf16* __restrict__ kb,
                                             __bf16* dst, int t0, int kvoff) {
  int lane = threadIdx.x & 31;
  __bf16* g0 = const_cast<__bf16*>(kb) + (size_t)(t0 + (lane >> 3)) * KVDIM + kvoff + (lane & 7) * 8;
  __bf16* l0 = dst + (lane >> 3) * 64 + (lane & 7) * 8;
#pragma unroll
  for (int i = 0; i < 16; ++i) {
#if HAVE_ASYNC_LDS
    __builtin_amdgcn_global_load_async_to_lds_b128(
        (gbl_v4i_ptr)(g0 + (size_t)i * 4 * KVDIM),
        (lds_v4i_ptr)(l0 + i * 4 * 64), 0, 0);
#else
    *(v8bf*)(l0 + i * 4 * 64) = *(const v8bf*)(g0 + (size_t)i * 4 * KVDIM);
#endif
  }
}

// ---------- kernels ----------

__global__ void cvt_bf16_kernel(const float* __restrict__ in, __bf16* __restrict__ out, int n) {
  int i = blockIdx.x * blockDim.x + threadIdx.x;
  if (i < n) out[i] = f2bf(in[i]);
}

// C[M,N] = A[M,K] @ Bt[N,K]^T    (one wave per 32x32 tile, 2x2 register blocking)
// mode 0: bf16 row-major C; mode 1: bf16 transposed C^T[N,M]; mode 2: f32 row-major C
__global__ __launch_bounds__(32) void gemm_wmma_kernel(
    const __bf16* __restrict__ A, const __bf16* __restrict__ Bt, void* __restrict__ Cout,
    int M, int N, int K, int mode) {
  int m0 = blockIdx.x * 32, n0 = blockIdx.y * 32;
  v8f acc[2][2] = {};
  for (int k0 = 0; k0 < K; k0 += 32) {
    v16bf a0 = load_a16x32(A, K, m0,      k0);
    v16bf a1 = load_a16x32(A, K, m0 + 16, k0);
    v16bf b0 = load_b32x16(Bt, K, n0,      k0);
    v16bf b1 = load_b32x16(Bt, K, n0 + 16, k0);
    acc[0][0] = wmma_bf16(a0, b0, acc[0][0]);
    acc[0][1] = wmma_bf16(a0, b1, acc[0][1]);
    acc[1][0] = wmma_bf16(a1, b0, acc[1][0]);
    acc[1][1] = wmma_bf16(a1, b1, acc[1][1]);
  }
  int lane = threadIdx.x & 31;
  int rbase = (lane >> 4) << 3;
  int nl = lane & 15;
#pragma unroll
  for (int mi = 0; mi < 2; ++mi) {
#pragma unroll
    for (int ni = 0; ni < 2; ++ni) {
      int mb = m0 + mi * 16 + rbase, nb = n0 + ni * 16 + nl;
      if (mode == 2) {
        float* C = (float*)Cout;
#pragma unroll
        for (int r = 0; r < 8; ++r) C[(size_t)(mb + r) * N + nb] = acc[mi][ni][r];
      } else if (mode == 1) {
        __bf16* C = (__bf16*)Cout;
#pragma unroll
        for (int r = 0; r < 8; ++r) C[(size_t)nb * M + mb + r] = f2bf(acc[mi][ni][r]);
      } else {
        __bf16* C = (__bf16*)Cout;
#pragma unroll
        for (int r = 0; r < 8; ++r) C[(size_t)(mb + r) * N + nb] = f2bf(acc[mi][ni][r]);
      }
    }
  }
}

// Flash attention: grid (S/16, H); 4 waves/block, wave = kv head; 64-wide t-chunks.
// q_bf [S, 1024], k_bf [S, 256], vT_bf [256, S]; output o_bf [S, 1024] (already /l /KV summed).
__global__ __launch_bounds__(128) void flash_attn_kernel(
    const __bf16* __restrict__ qb, const __bf16* __restrict__ kb,
    const __bf16* __restrict__ vtb, __bf16* __restrict__ ob) {
  __shared__ float osum[NKV][16][HD];                     // 16 KB combine buffer
  __shared__ __align__(16) __bf16 pst[NKV][16][64];       //  8 KB P staging (per wave)
  __shared__ __align__(16) __bf16 kst[NKV][2][64][64];    // 64 KB double-buffered K tiles

  const int wave = threadIdx.x >> 5;
  const int lane = threadIdx.x & 31;
  const int hi   = lane >> 4;
  const int nl   = lane & 15;
  const int s0   = blockIdx.x * 16;
  const int h    = blockIdx.y;
  const int kv   = wave;
  const float NEGINF = -__builtin_inff();

  // Q fragments for this head (K = 64 -> two 16x32 A frags)
  v16bf qa0 = load_a16x32(qb, QDIM, s0, h * HD);
  v16bf qa1 = load_a16x32(qb, QDIM, s0, h * HD + 32);

  float mrow[8], lrow[8];
  v8f oa[4] = {};
#pragma unroll
  for (int r = 0; r < 8; ++r) { mrow[r] = NEGINF; lrow[r] = 0.f; }

  int buf = 0;
  stage_k_tile(kb, &kst[kv][0][0][0], 0, kv * HD);
  wait_async_lds();

  for (int t0 = 0; t0 < s0 + 16; t0 += 64) {
    // prefetch next K chunk into the other buffer (overlaps with compute below)
    int nt0 = t0 + 64;
    bool pref = nt0 < s0 + 16;
    if (pref) stage_k_tile(kb, &kst[kv][buf ^ 1][0][0], nt0, kv * HD);

    const __bf16* kt = &kst[kv][buf][0][0];
    int nsub = (s0 + 16 - t0 + 15) >> 4; if (nsub > 4) nsub = 4;

    // scores: up to four 16x16 subtiles over t, K over d = 64 (uniform skip of masked subtiles)
    v8f sc[4] = {};
#pragma unroll
    for (int sub = 0; sub < 4; ++sub) {
      if (sub < nsub) {
        sc[sub] = wmma_bf16(qa0, load_b32x16(kt, 64, sub * 16, 0),  sc[sub]);
        sc[sub] = wmma_bf16(qa1, load_b32x16(kt, 64, sub * 16, 32), sc[sub]);
      }
    }

    // online softmax over the 64-wide chunk
#pragma unroll
    for (int r = 0; r < 8; ++r) {
      int srow = s0 + r + (hi << 3);
      int row  = r + (hi << 3);
      float p[4];
      float cm = NEGINF;
#pragma unroll
      for (int sub = 0; sub < 4; ++sub) {
        float a = sc[sub][r] * 0.125f;                // / sqrt(64)
        a = (t0 + sub * 16 + nl > srow) ? NEGINF : a; // causal mask
        p[sub] = a;
        cm = fmaxf(cm, a);
      }
      cm = rowmax16(cm);
      float mn = fmaxf(mrow[r], cm);
      float scale = __expf(mrow[r] - mn);
      float rs = 0.f;
#pragma unroll
      for (int sub = 0; sub < 4; ++sub) { p[sub] = __expf(p[sub] - mn); rs += p[sub]; }
      lrow[r] = lrow[r] * scale + rowsum16(rs);
      mrow[r] = mn;
#pragma unroll
      for (int dt = 0; dt < 4; ++dt) oa[dt][r] *= scale;
#pragma unroll
      for (int sub = 0; sub < 4; ++sub)
        pst[kv][row][sub * 16 + nl] = f2bf(p[sub]);
    }

    // P @ V (reload P in A-layout; same-wave LDS ops are in-order)
    int nkp = (nsub + 1) >> 1;
#pragma unroll
    for (int kp = 0; kp < 2; ++kp) {
      if (kp < nkp) {
        v16bf pa = load_a16x32(&pst[kv][0][0], 64, 0, kp * 32);
#pragma unroll
        for (int dt = 0; dt < 4; ++dt) {
          v16bf b = load_b32x16(vtb, S_LEN, kv * HD + dt * 16, t0 + kp * 32);
          oa[dt] = wmma_bf16(pa, b, oa[dt]);
        }
      }
    }

    if (pref) wait_async_lds();   // next chunk staged before we flip buffers
    buf ^= 1;
  }

  // normalize (softmax denom + /KV) and deposit to combine buffer
#pragma unroll
  for (int r = 0; r < 8; ++r) {
    float inv = 1.0f / (lrow[r] * (float)NKV);
    int row = r + (hi << 3);
#pragma unroll
    for (int dt = 0; dt < 4; ++dt)
      osum[kv][row][dt * 16 + nl] = oa[dt][r] * inv;
  }
  __syncthreads();

  for (int e = threadIdx.x; e < 16 * HD; e += 128) {
    int row = e >> 6, col = e & 63;
    float sum = osum[0][row][col] + osum[1][row][col] + osum[2][row][col] + osum[3][row][col];
    ob[(size_t)(s0 + row) * QDIM + h * HD + col] = f2bf(sum);
  }
}

// ---------- launcher ----------

extern "C" void kernel_launch(void* const* d_in, const int* in_sizes, int n_in,
                              void* d_out, int out_size, void* d_ws, size_t ws_size,
                              hipStream_t stream) {
  const float* x  = (const float*)d_in[0];
  const float* Wq = (const float*)d_in[1];
  const float* Wk = (const float*)d_in[2];
  const float* Wv = (const float*)d_in[3];
  const float* Wo = (const float*)d_in[4];

  char* ws = (char*)d_ws;
  const size_t MB = 1024u * 1024u;
  __bf16* xb  = (__bf16*)(ws + 0);                    // [2048, 512]   2 MB
  __bf16* wqb = (__bf16*)(ws + 2 * MB);               // [1024, 512]   1 MB
  __bf16* wkb = (__bf16*)(ws + 3 * MB);               // [256, 512]    256 KB
  __bf16* wvb = (__bf16*)(ws + 3 * MB + 256 * 1024);  // [256, 512]    256 KB
  __bf16* wob = (__bf16*)(ws + 3 * MB + 512 * 1024);  // [512, 1024]   1 MB
  __bf16* qbuf  = (__bf16*)(ws + 4 * MB + 512 * 1024);   // [2048, 1024]  4 MB
  __bf16* kbuf  = (__bf16*)(ws + 8 * MB + 512 * 1024);   // [2048, 256]   1 MB
  __bf16* vtbuf = (__bf16*)(ws + 9 * MB + 512 * 1024);   // [256, 2048]   1 MB (transposed)
  __bf16* obuf  = (__bf16*)(ws + 10 * MB + 512 * 1024);  // [2048, 1024]  4 MB

  // convert inputs to bf16
  cvt_bf16_kernel<<<(S_LEN * IN_DIM + 255) / 256, 256, 0, stream>>>(x,  xb,  S_LEN * IN_DIM);
  cvt_bf16_kernel<<<(QDIM * IN_DIM + 255) / 256, 256, 0, stream>>>(Wq, wqb, QDIM * IN_DIM);
  cvt_bf16_kernel<<<(KVDIM * IN_DIM + 255) / 256, 256, 0, stream>>>(Wk, wkb, KVDIM * IN_DIM);
  cvt_bf16_kernel<<<(KVDIM * IN_DIM + 255) / 256, 256, 0, stream>>>(Wv, wvb, KVDIM * IN_DIM);
  cvt_bf16_kernel<<<(OUT_DIM * QDIM + 255) / 256, 256, 0, stream>>>(Wo, wob, OUT_DIM * QDIM);

  // projections: q = x @ Wq^T, k = x @ Wk^T, v^T = (x @ Wv^T)^T
  gemm_wmma_kernel<<<dim3(S_LEN / 32, QDIM / 32),  32, 0, stream>>>(xb, wqb, qbuf,  S_LEN, QDIM,  IN_DIM, 0);
  gemm_wmma_kernel<<<dim3(S_LEN / 32, KVDIM / 32), 32, 0, stream>>>(xb, wkb, kbuf,  S_LEN, KVDIM, IN_DIM, 0);
  gemm_wmma_kernel<<<dim3(S_LEN / 32, KVDIM / 32), 32, 0, stream>>>(xb, wvb, vtbuf, S_LEN, KVDIM, IN_DIM, 1);

  // causal attention per (kv, h), averaged over kv
  flash_attn_kernel<<<dim3(S_LEN / 16, NH), 128, 0, stream>>>(qbuf, kbuf, vtbuf, obuf);

  // output projection: out = o @ Wo^T (f32)
  gemm_wmma_kernel<<<dim3(S_LEN / 32, OUT_DIM / 32), 32, 0, stream>>>(obuf, wob, d_out, S_LEN, OUT_DIM, QDIM, 2);
}
